// loss_A2C_pair_70411693851036
// MI455X (gfx1250) — matrix operand order repaired and checked by reading, not compiled, not run on hardware
//
#include <hip/hip_runtime.h>
#include <cstdint>

#define BATCH 4
#define CHN   256
#define IH    256
#define IW    256
#define NCHUNK 4
#define CPC   (CHN/NCHUNK)    // 64 channels per workgroup
#define TILE_H 64
#define TILE_W 32
#define REG_H (TILE_H+4)      // 68
#define REG_W (TILE_W+4)      // 36
#define REG_N (REG_H*REG_W)   // 2448
#define HB_STRIDE 24
#define HB_SIZE (16*HB_STRIDE + 16) // 400 halves, slack absorbs dead K>=20 reads

typedef __attribute__((ext_vector_type(16))) _Float16 v16h;
typedef __attribute__((ext_vector_type(8)))  float    v8f;

struct U32x8 { uint32_t u[8]; };
static_assert(sizeof(U32x8) == sizeof(v16h), "bitcast size");

// ---------------- Phase A: per-pixel flags ----------------
// bit0 corr0, bit1 corr1, bit2 edge, bit3 cm0, bit4 cm1, bit5 pred==1
__global__ void phaseA(const int* __restrict__ gts, const float* __restrict__ clsf,
                       uint8_t* __restrict__ mask) {
  int i = blockIdx.x * 256 + threadIdx.x;      // < B*H*W
  int g = gts[i];
  int b = i >> 16;
  int rem = i & 65535;
  int h = rem >> 8, w = rem & 255;
  const float* cp = clsf + ((size_t)(b * 2) << 16);
  int pred1 = cp[65536 + rem] > cp[rem];       // argmax ties -> class 0
  int ud = (h < IH - 5) && (gts[i + 5 * IW] != g);
  int lr = (w < IW - 5) && (gts[i + 5] != g);
  int e = ud | lr | (g == -1);
  uint8_t m = 0;
  if ((g == 0) & (pred1 == 0)) m |= 1;
  if ((g == 1) & (pred1 == 1)) m |= 2;
  if (e)        m |= 4;
  if (g == 0)   m |= 8;
  if (g == 1)   m |= 16;
  if (pred1)    m |= 32;
  mask[i] = m;
}

// ---------------- Phase B: ring counts, selectors, valid counters ----------------
__global__ void phaseB(const uint8_t* __restrict__ mask, uint8_t* __restrict__ sel,
                       float* __restrict__ rcp, unsigned int* __restrict__ hdr) {
  __shared__ unsigned int cred[4];
  if (threadIdx.x < 4) cred[threadIdx.x] = 0;
  __syncthreads();

  int i = blockIdx.x * 256 + threadIdx.x;
  int b = i >> 16; int rem = i & 65535; int h = rem >> 8, w = rem & 255;
  uint8_t mc = mask[i];
  int ccls0 = 0, ccls1 = 0, ccor0 = 0, ccor1 = 0;
  for (int dy = -2; dy <= 2; ++dy) {
    int hh = h + dy; if ((unsigned)hh >= IH) continue;
    for (int dx = -2; dx <= 2; ++dx) {
      int ww = w + dx; if ((unsigned)ww >= IW) continue;
      uint8_t mm = mask[(b << 16) + (hh << 8) + ww];
      ccor0 += mm & 1;        ccor1 += (mm >> 1) & 1;
      ccls0 += (mm >> 3) & 1; ccls1 += (mm >> 4) & 1;
    }
  }
  ccor0 -= mc & 1;        ccor1 -= (mc >> 1) & 1;   // ring = box - center
  ccls0 -= (mc >> 3) & 1; ccls1 -= (mc >> 4) & 1;
  int e = (mc >> 2) & 1, cm0 = (mc >> 3) & 1, cm1 = (mc >> 4) & 1, pm1 = (mc >> 5) & 1;
  int cal0 = (ccls0 >= 1) & e & cm0;
  int cal1 = (ccls1 >= 1) & e & cm1;
  int mse0 = (ccor0 >= 1) & e & cm0 & (pm1 ^ 1);
  int mse1 = (ccor1 >= 1) & e & cm1 & pm1;
  uint8_t s = mse0 ? 1 : (mse1 ? 2 : 0);
  sel[i] = s;
  float cc = mse0 ? (float)ccor0 : (float)ccor1;
  rcp[i] = s ? 1.0f / (cc + 1e-5f) : 0.0f;

  if (cal0) atomicAdd(&cred[0], 1u);
  if (cal1) atomicAdd(&cred[1], 1u);
  if (mse0) atomicAdd(&cred[2], 1u);
  if (mse1) atomicAdd(&cred[3], 1u);
  __syncthreads();
  if (threadIdx.x < 4) atomicAdd(&hdr[2 + threadIdx.x], cred[threadIdx.x]);
}

// ---------------- Phase C helpers ----------------
// A operand (16-bit A 16x32 layout): lanes0-15 row m=lane, K in {0..7,16..23};
// lanes16-31 row m=lane-16, K in {8..15,24..31}. K>=20 never loaded (band is 0 there).
__device__ __forceinline__ v16h loadA(const _Float16* gp, int rr, int cb, bool hi) {
  U32x8 t;
  const int kb0 = hi ? 8 : 0;
  const int kb1 = hi ? 24 : 16;
  const int base = rr * REG_W + cb;
#pragma unroll
  for (int j = 0; j < 4; ++j)
    t.u[j] = *reinterpret_cast<const uint32_t*>(gp + base + kb0 + 2 * j);
#pragma unroll
  for (int j = 0; j < 4; ++j) {
    int kp = kb1 + 2 * j;
    t.u[4 + j] = (kp <= 18) ? *reinterpret_cast<const uint32_t*>(gp + base + kp) : 0u;
  }
  return __builtin_bit_cast(v16h, t);
}

// Full 5x5 box sum over one 16x16 subtile: 2 horizontal + 1 vertical banded WMMA.
__device__ __forceinline__ v8f computeBox(const _Float16* gp, _Float16* hb,
                                          v16h bandA, v16h bandB,
                                          int rowS, int cb, int lane) {
  const int mn = lane & 15;
  const bool hi = lane >= 16;
  v8f cz = {};
  // horizontal, rows rowS..rowS+15 (region rows row0-2..row0+13)
  {
    v16h a = loadA(gp, rowS + mn, cb, hi);
    v8f h0 = __builtin_amdgcn_wmma_f32_16x16x32_f16(false, a, false, bandB,
                                                    (short)0, cz, false, false);
#pragma unroll
    for (int r = 0; r < 8; ++r)
      hb[mn * HB_STRIDE + r + (hi ? 8 : 0)] = (_Float16)h0[r];
  }
  // horizontal, rows rowS+16..rowS+19 (only output rows 0..3 kept)
  {
    int rr = rowS + 16 + mn; if (rr > REG_H - 1) rr = REG_H - 1;
    v16h a = loadA(gp, rr, cb, hi);
    v8f h1 = __builtin_amdgcn_wmma_f32_16x16x32_f16(false, a, false, bandB,
                                                    (short)0, cz, false, false);
    if (!hi) {
#pragma unroll
      for (int r = 0; r < 4; ++r)
        hb[mn * HB_STRIDE + 16 + r] = (_Float16)h1[r];
    }
  }
  __syncthreads();   // order H stores before B-operand loads (uniform across block)
  // vertical: B operand (32x16 layout): lanes0-15 col=lane, K=2j,2j+1;
  // lanes16-31 col=lane-16, K=16+2j,2j+1 (K>=20 slots garbage, killed by bandA=0)
  U32x8 t;
  const int kb = hi ? 16 : 0;
#pragma unroll
  for (int j = 0; j < 8; ++j)
    t.u[j] = *reinterpret_cast<const uint32_t*>(hb + mn * HB_STRIDE + kb + 2 * j);
  v16h bv = __builtin_bit_cast(v16h, t);
  return __builtin_amdgcn_wmma_f32_16x16x32_f16(false, bandA, false, bv,
                                                (short)0, cz, false, false);
}

// ---------------- Phase C: heavy pass ----------------
__global__ __launch_bounds__(256) void phaseC(const float* __restrict__ feat,
                                              const uint8_t* __restrict__ mask,
                                              const uint8_t* __restrict__ sel,
                                              const float* __restrict__ rcp,
                                              float* __restrict__ losses) {
  __shared__ alignas(16) float    Fpl[REG_N];
  __shared__ alignas(16) _Float16 g0pl[REG_N];
  __shared__ alignas(16) _Float16 g1pl[REG_N];
  __shared__ uint8_t  mpl[REG_N];
  __shared__ alignas(16) _Float16 hb0[8 * HB_SIZE];
  __shared__ alignas(16) _Float16 hb1[8 * HB_SIZE];
  __shared__ float red[256];

  const int tid = threadIdx.x;
  const int lane = tid & 31;
  const int wv = tid >> 5;                 // 8 waves
  const int b   = blockIdx.x / NCHUNK;
  const int c0  = (blockIdx.x % NCHUNK) * CPC;
  const int th0 = blockIdx.y * TILE_H;
  const int tw0 = blockIdx.z * TILE_W;
  const int sy = wv >> 1, sx = wv & 1;     // 4x2 subtiles of 16x16
  const int mn = lane & 15;
  const bool hi = lane >= 16;

  // stage mask plane once per tile
  for (int i = tid; i < REG_N; i += 256) {
    int r = i / REG_W, c = i - r * REG_W;
    int gh = th0 - 2 + r, gw = tw0 - 2 + c;
    uint8_t m = 0;
    if ((unsigned)gh < IH && (unsigned)gw < IW) m = mask[(b << 16) + (gh << 8) + gw];
    mpl[i] = m;
  }

  // banded operands (exact 0/1 in f16). band(m,k)=1 iff k in [m,m+4], k<20.
  v16h bandB, bandA;
#pragma unroll
  for (int e = 0; e < 16; ++e) {
    int j = e >> 1, hh = e & 1;
    int kB = (j < 4) ? ((hi ? 8 : 0) + 2 * j + hh)
                     : ((hi ? 24 : 16) + 2 * (j - 4) + hh);
    bandB[e] = (_Float16)((kB >= mn && kB <= mn + 4 && kB < 20) ? 1.0f : 0.0f);
    int kA = (hi ? 16 : 0) + 2 * j + hh;
    bandA[e] = (_Float16)((kA >= mn && kA <= mn + 4 && kA < 20) ? 1.0f : 0.0f);
  }

  // channel-invariant per-pixel weights for this lane's 8 pixels (C-layout rows)
  float w0a[8], w1a[8], m0a[8], m1a[8];
#pragma unroll
  for (int r = 0; r < 8; ++r) {
    int row = r + (hi ? 8 : 0);
    int gh = th0 + 16 * sy + row;
    int gw = tw0 + 16 * sx + mn;
    int gi = (b << 16) + (gh << 8) + gw;
    uint8_t s = sel[gi];
    float rc = rcp[gi];
    w0a[r] = (s == 1) ? rc : 0.0f;  m0a[r] = (s == 1) ? 1.0f : 0.0f;
    w1a[r] = (s == 2) ? rc : 0.0f;  m1a[r] = (s == 2) ? 1.0f : 0.0f;
  }

  _Float16* myhb0 = hb0 + wv * HB_SIZE;
  _Float16* myhb1 = hb1 + wv * HB_SIZE;
  const int cb   = 16 * sx;   // region col of k=0 (= col0-2)
  const int rowS = 16 * sy;   // region row of first box input row (= row0-2)

  float acc0 = 0.f, acc1 = 0.f;
  const float* fbase = feat + (((size_t)b * CHN + c0) << 16);

  for (int ch = 0; ch < CPC; ++ch) {
    __syncthreads();                       // protect prior reads of LDS planes
    const float* fc = fbase + ((size_t)ch << 16);
    for (int i = tid; i < REG_N; i += 256) {
      int r = i / REG_W, c = i - r * REG_W;
      int gh = th0 - 2 + r, gw = tw0 - 2 + c;
      float f = 0.f;
      if ((unsigned)gh < IH && (unsigned)gw < IW) {
        const float* p = fc + (gh << 8) + gw;
        f = *p;
        if (ch + 1 < CPC) __builtin_prefetch(p + IH * IW, 0, 1);  // next channel
      }
      Fpl[i] = f;
      _Float16 h16 = (_Float16)f;
      uint8_t m = mpl[i];
      g0pl[i] = (m & 1) ? h16 : (_Float16)0.0f;   // F * corr0
      g1pl[i] = (m & 2) ? h16 : (_Float16)0.0f;   // F * corr1
    }
    __syncthreads();

    v8f S0 = computeBox(g0pl, myhb0, bandA, bandB, rowS, cb, lane);
    v8f S1 = computeBox(g1pl, myhb1, bandA, bandB, rowS, cb, lane);

#pragma unroll
    for (int r = 0; r < 8; ++r) {
      int row = r + (hi ? 8 : 0);
      int i = (rowS + 2 + row) * REG_W + (cb + 2 + mn);
      float f = Fpl[i];
      float ne0 = S0[r] - (float)g0pl[i];          // ring = box - center
      float ne1 = S1[r] - (float)g1pl[i];
      float d0 = m0a[r] * f - ne0 * w0a[r];        // m*(F - neigh*rcp), m in {0,1}
      float d1 = m1a[r] * f - ne1 * w1a[r];
      acc0 = fmaf(d0, d0, acc0);
      acc1 = fmaf(d1, d1, acc1);
    }
  }

  // block reduction + one atomic per class
  red[tid] = acc0; __syncthreads();
  for (int s = 128; s > 0; s >>= 1) { if (tid < s) red[tid] += red[tid + s]; __syncthreads(); }
  if (tid == 0) atomicAdd(&losses[0], red[0]);
  __syncthreads();
  red[tid] = acc1; __syncthreads();
  for (int s = 128; s > 0; s >>= 1) { if (tid < s) red[tid] += red[tid + s]; __syncthreads(); }
  if (tid == 0) atomicAdd(&losses[1], red[0]);
}

// ---------------- Phase D: scalar epilogue ----------------
__global__ void phaseD(const float* __restrict__ losses,
                       const unsigned int* __restrict__ cnts, float* __restrict__ out) {
  float total = 0.f; int ncls = 0;
  for (int c = 0; c < 2; ++c) {
    unsigned ncal = cnts[c], nmse = cnts[2 + c];
    float lc = losses[c] / fmaxf((float)nmse * (float)CHN, 1.0f);
    bool valid = (ncal >= 1) && (nmse >= 1);
    if (valid) { total += lc; ++ncls; }
  }
  out[0] = ncls ? total / (float)ncls : total;
}

extern "C" void kernel_launch(void* const* d_in, const int* in_sizes, int n_in,
                              void* d_out, int out_size, void* d_ws, size_t ws_size,
                              hipStream_t stream) {
  (void)in_sizes; (void)n_in; (void)out_size; (void)ws_size;
  const float* feat = (const float*)d_in[0];
  const float* clsf = (const float*)d_in[1];
  const int*   gts  = (const int*)d_in[2];

  uint8_t* ws = (uint8_t*)d_ws;
  float*        losses  = (float*)ws;             // [0..1]
  unsigned int* hdr     = (unsigned int*)ws;      // [2..5] = ncal0,ncal1,nmse0,nmse1
  uint8_t*      maskbuf = ws + 64;
  uint8_t*      selbuf  = ws + 64 + 262144;
  float*        rcpbuf  = (float*)(ws + 64 + 2 * 262144);

  hipMemsetAsync(d_ws, 0, 64, stream);

  const int npx = BATCH * IH * IW;                // 262144
  phaseA<<<npx / 256, 256, 0, stream>>>(gts, clsf, maskbuf);
  phaseB<<<npx / 256, 256, 0, stream>>>(maskbuf, selbuf, rcpbuf, hdr);
  dim3 grid(BATCH * NCHUNK, IH / TILE_H, IW / TILE_W);   // (16,4,8) = 512 WGs
  phaseC<<<grid, 256, 0, stream>>>(feat, maskbuf, selbuf, rcpbuf, losses);
  phaseD<<<1, 1, 0, stream>>>(losses, hdr + 2, (float*)d_out);
}
